// META_RNNDetector_9500467659341
// MI455X (gfx1250) — matrix operand order repaired and checked by reading, not compiled, not run on hardware
//
#include <hip/hip_runtime.h>

typedef __attribute__((ext_vector_type(16))) _Float16 v16h;
typedef __attribute__((ext_vector_type(8)))  float    v8f;

#define HID    256
#define K0EXT  288   // layer-0 K: 256 (h0) + 4 (x window) + 28 zero pad
#define BROWS  16
#define NTHR   512
#define HPAD0  296   // row stride 592B = 148 dw == 20 mod 64 -> conflict-free b128
#define HPAD1  264   // row stride 528B = 132 dw ==  4 mod 64 -> conflict-free b128

struct Frag { union { v16h v; uint4 u[2]; }; };

// B fragment (32x16 f16): lanes 0-15 hold K=k0..k0+15 of column n, lanes 16-31 K=k0+16..31.
__device__ __forceinline__ v16h load_b_frag(const _Float16* wp, int k0) {
  Frag f;
  const _Float16* p = wp + k0;
  f.u[0] = *(const uint4*)(p);
  f.u[1] = *(const uint4*)(p + 8);
  return f.v;
}

// A fragment (16x32 f16) from LDS h buffer with row pitch `pitch`.
__device__ __forceinline__ v16h load_a_frag(const _Float16* hbuf, int pitch, int k0, int lane) {
  const int m   = lane & 15;
  const int sel = (lane >> 4) << 4;
  const _Float16* p = hbuf + m * pitch + k0 + sel;
  Frag f;
  f.u[0] = *(const uint4*)(p);        // K = k0+sel .. +7
  f.u[1] = *(const uint4*)(p + 8);    // K = k0+sel+8 .. +15
  return f.v;
}

// Software-pipelined 4-tile GEMM chain: prefetch (k+32) A/B while issuing k's WMMAs.
template<int KLEN, int PITCH>
__device__ __forceinline__ void gemm4(v8f acc[4], const _Float16* abuf,
                                      const _Float16* wbase, const int* hoff, int lane) {
  v16h ac = load_a_frag(abuf, PITCH, 0, lane);
  v16h bc[4];
  #pragma unroll
  for (int g = 0; g < 4; ++g) bc[g] = load_b_frag(wbase + hoff[g], 0);
  #pragma unroll
  for (int k0 = 0; k0 < KLEN; k0 += 32) {
    v16h an = ac;
    v16h bn[4];
    if (k0 + 32 < KLEN) {
      an = load_a_frag(abuf, PITCH, k0 + 32, lane);
      #pragma unroll
      for (int g = 0; g < 4; ++g) bn[g] = load_b_frag(wbase + hoff[g], k0 + 32);
    } else {
      #pragma unroll
      for (int g = 0; g < 4; ++g) bn[g] = bc[g];
    }
    #pragma unroll
    for (int g = 0; g < 4; ++g)
      acc[g] = __builtin_amdgcn_wmma_f32_16x16x32_f16(
          false, ac, false, bc[g], (short)0, acc[g], false, false);
    ac = an;
    #pragma unroll
    for (int g = 0; g < 4; ++g) bc[g] = bn[g];
  }
}

__device__ __forceinline__ float sigmoid_fast(float x) {
  return __builtin_amdgcn_rcpf(1.f + __builtin_amdgcn_exp2f(-1.44269504f * x));
}
__device__ __forceinline__ float tanh_fast(float x) {
  return 1.f - 2.f * __builtin_amdgcn_rcpf(1.f + __builtin_amdgcn_exp2f(2.88539008f * x));
}

// ---- prep: pack f16 weight layouts + fold biases (one-shot, fully parallel) ----
__global__ void prep_weights(const float* __restrict__ Whh0, const float* __restrict__ Wih0,
                             const float* __restrict__ Wih1, const float* __restrict__ Whh1,
                             const float* __restrict__ Wfc,
                             const float* __restrict__ bih0, const float* __restrict__ bhh0,
                             const float* __restrict__ bih1, const float* __restrict__ bhh1,
                             _Float16* __restrict__ W0c, _Float16* __restrict__ W1h,
                             _Float16* __restrict__ W2h, _Float16* __restrict__ Wfc16,
                             float* __restrict__ bias0, float* __restrict__ bias1) {
  const int i = blockIdx.x * blockDim.x + threadIdx.x;
  if (i < 1024 * K0EXT) {                 // combined [W_hh0 | W_ih0 | 0] as [1024][288]
    const int row = i / K0EXT, col = i % K0EXT;
    float v = 0.f;
    if (col < HID)            v = Whh0[row * HID + col];
    else if (col < HID + 4)   v = Wih0[row * 4 + (col - HID)];
    W0c[i] = (_Float16)v;
  }
  if (i < 1024 * HID) { W1h[i] = (_Float16)Wih1[i]; W2h[i] = (_Float16)Whh1[i]; }
  if (i < 16 * HID) {                     // W_fc padded to 16 output columns
    const int row = i / HID, col = i % HID;
    Wfc16[i] = (_Float16)(row < 2 ? Wfc[row * HID + col] : 0.f);
  }
  if (i < 1024) { bias0[i] = bih0[i] + bhh0[i]; bias1[i] = bih1[i] + bhh1[i]; }
}

// ---- persistent 2-layer LSTM: block = 16 batch rows, 16 waves; wave w owns gate
// columns [16w,16w+16) of i,f,g,o -> wave-local cell update, c-state in VGPRs. ----
__global__ __launch_bounds__(NTHR)
void lstm2_persistent(const float* __restrict__ y,
                      const float* __restrict__ bfc,
                      const _Float16* __restrict__ W0c,
                      const _Float16* __restrict__ W1h,
                      const _Float16* __restrict__ W2h,
                      const _Float16* __restrict__ Wfc16,
                      const float* __restrict__ bias0,
                      const float* __restrict__ bias1,
                      float* __restrict__ out, int T) {
  __shared__ _Float16 h0buf[2][BROWS * HPAD0];  // cols 0-255: h0; 256-259: x window; rest 0
  __shared__ _Float16 h1buf[2][BROWS * HPAD1];

  const int tid  = threadIdx.x;
  const int lane = tid & 31;
  const int wv   = tid >> 5;
  const int bb   = blockIdx.x * BROWS;

  for (int i = tid; i < BROWS * HPAD0; i += NTHR) { h0buf[0][i] = (_Float16)0.f; h0buf[1][i] = (_Float16)0.f; }
  for (int i = tid; i < BROWS * HPAD1; i += NTHR) { h1buf[0][i] = (_Float16)0.f; h1buf[1][i] = (_Float16)0.f; }

  const int j  = (wv << 4) + (lane & 15);  // gate-chunk column owned by this lane
  const int kh = (lane >> 4) << 4;         // B-fragment K-half (0 / 16)
  const int m0 = (lane >> 4) << 3;         // C/D fragment row base (0 / 8)

  int hoff0[4], hoff[4];
  float b0[4], b1[4];
  #pragma unroll
  for (int g = 0; g < 4; ++g) {
    const int n = (g << 8) + j;
    hoff0[g] = n * K0EXT + kh;
    hoff[g]  = n * HID + kh;
    b0[g] = bias0[n];
    b1[g] = bias1[n];
  }
  const int   hoffF  = (lane & 15) * HID + kh;
  const float fcinit = ((lane & 15) < 2) ? bfc[lane & 15] : 0.f;

  float c0r[8], c1r[8];
  #pragma unroll
  for (int r = 0; r < 8; ++r) { c0r[r] = 0.f; c1r[r] = 0.f; }

  // y sliding window (wave 0, lane m only): y[t-3], y[t-2], y[t-1]
  float yw0 = -100.f, yw1 = -100.f, yw2 = -100.f;

  int p = 0;
  for (int t = 0; t < T; ++t) {
    const int q = p ^ 1;

    // Opaque zero per iteration: keeps the per-step weight streams (1.7MB from L2)
    // from being LICM-hoisted out of the t loop and spilled to scratch.
    int opq;
    asm volatile("s_mov_b32 %0, 0" : "=s"(opq));
    const _Float16* W0t = W0c   + opq;
    const _Float16* W1t = W1h   + opq;
    const _Float16* W2t = W2h   + opq;
    const _Float16* WfT = Wfc16 + opq;

    if (wv == 0 && lane < BROWS) {
      const float ycur = y[(size_t)(bb + lane) * T + t];
      union { uint2 u; _Float16 h[4]; } pk;
      pk.h[0] = (_Float16)yw0; pk.h[1] = (_Float16)yw1;
      pk.h[2] = (_Float16)yw2; pk.h[3] = (_Float16)ycur;
      *(uint2*)&h0buf[p][lane * HPAD0 + HID] = pk.u;   // x -> A columns 256..259
      yw0 = yw1; yw1 = yw2; yw2 = ycur;
    }
    __syncthreads();   // x ready; also separates prev step's h1 writes

    // ---------------- layer 0: gates = [h0|x] @ [Whh0|Wih0]^T + b0 ----------------
    v8f acc[4];
    #pragma unroll
    for (int g = 0; g < 4; ++g)
      #pragma unroll
      for (int r = 0; r < 8; ++r) acc[g][r] = b0[g];
    gemm4<K0EXT, HPAD0>(acc, h0buf[p], W0t, hoff0, lane);
    #pragma unroll
    for (int r = 0; r < 8; ++r) {
      const float iv = sigmoid_fast(acc[0][r]);
      const float fv = sigmoid_fast(acc[1][r]);
      const float gv = tanh_fast(acc[2][r]);
      const float ov = sigmoid_fast(acc[3][r]);
      const float c  = fv * c0r[r] + iv * gv;
      c0r[r] = c;
      h0buf[q][(m0 + r) * HPAD0 + j] = (_Float16)(ov * tanh_fast(c));
    }
    __syncthreads();   // h0_new visible to all waves

    // ---------------- layer 1: gates = h0_new @ Wih1^T + h1 @ Whh1^T + b1 ----------------
    #pragma unroll
    for (int g = 0; g < 4; ++g)
      #pragma unroll
      for (int r = 0; r < 8; ++r) acc[g][r] = b1[g];
    gemm4<HID, HPAD0>(acc, h0buf[q], W1t, hoff, lane);
    gemm4<HID, HPAD1>(acc, h1buf[p], W2t, hoff, lane);
    #pragma unroll
    for (int r = 0; r < 8; ++r) {
      const float iv = sigmoid_fast(acc[0][r]);
      const float fv = sigmoid_fast(acc[1][r]);
      const float gv = tanh_fast(acc[2][r]);
      const float ov = sigmoid_fast(acc[3][r]);
      const float c  = fv * c1r[r] + iv * gv;
      c1r[r] = c;
      h1buf[q][(m0 + r) * HPAD1 + j] = (_Float16)(ov * tanh_fast(c));
    }
    __syncthreads();   // h1_new visible

    // ---------------- FC head on the matrix pipe (wave 0): out = h1 @ Wfc16^T + bfc ----
    if (wv == 0) {
      v8f af;
      #pragma unroll
      for (int r = 0; r < 8; ++r) af[r] = fcinit;
      v16h ac = load_a_frag(h1buf[q], HPAD1, 0, lane);
      v16h bc = load_b_frag(WfT + hoffF, 0);
      #pragma unroll
      for (int k0 = 0; k0 < HID; k0 += 32) {
        v16h an = ac, bn = bc;
        if (k0 + 32 < HID) {
          an = load_a_frag(h1buf[q], HPAD1, k0 + 32, lane);
          bn = load_b_frag(WfT + hoffF, k0 + 32);
        }
        af = __builtin_amdgcn_wmma_f32_16x16x32_f16(
            false, ac, false, bc, (short)0, af, false, false);
        ac = an; bc = bn;
      }
      if ((lane & 15) < 2) {
        const int cls = lane & 15;
        #pragma unroll
        for (int r = 0; r < 8; ++r)
          out[((size_t)(bb + m0 + r) * T + t) * 2 + cls] = af[r];
      }
    }
    p = q;
  }
}

extern "C" void kernel_launch(void* const* d_in, const int* in_sizes, int n_in,
                              void* d_out, int out_size, void* d_ws, size_t ws_size,
                              hipStream_t stream) {
  const float* y    = (const float*)d_in[0];
  const float* Wih0 = (const float*)d_in[1];
  const float* Whh0 = (const float*)d_in[2];
  const float* bih0 = (const float*)d_in[3];
  const float* bhh0 = (const float*)d_in[4];
  const float* Wih1 = (const float*)d_in[5];
  const float* Whh1 = (const float*)d_in[6];
  const float* bih1 = (const float*)d_in[7];
  const float* bhh1 = (const float*)d_in[8];
  const float* Wfc  = (const float*)d_in[9];
  const float* bfc  = (const float*)d_in[10];
  float* out = (float*)d_out;

  const int B = 64;
  const int T = in_sizes[0] / B;

  char* ws = (char*)d_ws;
  _Float16* W0c   = (_Float16*)(ws + 0);        // 1024*288*2 = 589824
  _Float16* W1h   = (_Float16*)(ws + 589824);   // 524288
  _Float16* W2h   = (_Float16*)(ws + 1114112);  // 524288
  _Float16* Wfc16 = (_Float16*)(ws + 1638400);  // 8192
  float*    bias0 = (float*)(ws + 1646592);     // 4096
  float*    bias1 = (float*)(ws + 1650688);     // 4096

  prep_weights<<<1152, 256, 0, stream>>>(Whh0, Wih0, Wih1, Whh1, Wfc,
                                         bih0, bhh0, bih1, bhh1,
                                         W0c, W1h, W2h, Wfc16, bias0, bias1);
  lstm2_persistent<<<B / BROWS, NTHR, 0, stream>>>(y, bfc, W0c, W1h, W2h, Wfc16,
                                                   bias0, bias1, out, T);
}
